// DNN_Beamformer_46093589020854
// MI455X (gfx1250) — matrix operand order repaired and checked by reading, not compiled, not run on hardware
//
#include <hip/hip_runtime.h>
#include <math.h>

// Problem constants (match reference)
#define B_    16
#define F_    257
#define C_    8
#define T_    1024
#define ATT_  320
#define KPAD_ 288          // 257 padded to multiple of 32 for f16 WMMA K
#define M_    (B_ * C_)    // 128 rows of the attention GEMM
#define BF_   (B_ * F_)    // 4112 independent (b,f) problems
#define EPS_  1e-15f

typedef __attribute__((ext_vector_type(16))) _Float16 v16h;
typedef __attribute__((ext_vector_type(8)))  _Float16 v8h;
typedef __attribute__((ext_vector_type(8)))  float    v8f;

// ---------------------------------------------------------------------------
// Kernel 0: w_mlp (257x320 f32, K-major) -> f16 TRANSPOSED (320 x 288pad,
// N-major). Each WMMA lane's B fragment then becomes two 16-byte runs.
// ---------------------------------------------------------------------------
__global__ void k_conv_wmlp(const float* __restrict__ wm, _Float16* __restrict__ wm16t) {
    int i = blockIdx.x * blockDim.x + threadIdx.x;
    if (i >= ATT_ * KPAD_) return;
    int n = i / KPAD_, k = i % KPAD_;
    float v = (k < F_) ? wm[k * ATT_ + n] : 0.0f;
    wm16t[i] = (_Float16)v;
}

// ---------------------------------------------------------------------------
// Kernel 1: psd_feat(b,c,f) = | mean_{d!=c} psd_s[b,f,c,d] | -> f16, padded K
// ---------------------------------------------------------------------------
__global__ void k_feat(const float* __restrict__ sr, const float* __restrict__ si,
                       _Float16* __restrict__ feat) {
    int i = blockIdx.x * blockDim.x + threadIdx.x;
    if (i >= M_ * KPAD_) return;
    int m = i / KPAD_, k = i % KPAD_;
    float out = 0.0f;
    if (k < F_) {
        int b = m >> 3, c = m & 7;
        size_t base = ((size_t)(b * F_ + k) * C_ + c) * C_;
        float ar = 0.f, ai = 0.f;
        #pragma unroll
        for (int d = 0; d < C_; ++d) {
            if (d != c) { ar += sr[base + d]; ai += si[base + d]; }
        }
        ar *= (1.0f / (C_ - 1)); ai *= (1.0f / (C_ - 1));
        out = sqrtf(ar * ar + ai * ai);
    }
    feat[i] = (_Float16)out;
}

// ---------------------------------------------------------------------------
// Global abs-max reduction for temp = max(|psd_n.real|, |psd_n.imag|)
// (reference reduces over ALL axes -> single scalar). Values >= 0, so
// float bits are monotone as unsigned -> atomicMax on uint works.
// ---------------------------------------------------------------------------
__global__ void k_init_temp(unsigned* t) {
    if (blockIdx.x == 0 && threadIdx.x == 0) *t = 0u;
}

__global__ void k_absmax(const float* __restrict__ a, int n, int addDiagEps,
                         unsigned* __restrict__ tmax) {
    float m = 0.0f;
    for (int i = blockIdx.x * blockDim.x + threadIdx.x; i < n;
         i += gridDim.x * blockDim.x) {
        float v = a[i];
        if (addDiagEps) {                       // EPS added to real diagonal first
            int c = (i >> 3) & 7, d = i & 7;
            if (c == d) v += EPS_;
        }
        m = fmaxf(m, fabsf(v));
    }
    #pragma unroll
    for (int off = 16; off > 0; off >>= 1) m = fmaxf(m, __shfl_down(m, off));
    if ((threadIdx.x & 31) == 0) atomicMax(tmax, __float_as_uint(m));
}

// ---------------------------------------------------------------------------
// Kernel 2: WMMA GEMM  h = tanh(feat(128xK) @ w_mlp(Kx320) + b_mlp)
// One wave per 16x16 output tile; 9 unrolled v_wmma_f32_16x16x32_f16 steps.
// Both fragments load as contiguous b128 pairs (A row-major, B from the
// transposed copy). Fragment layouts per CDNA5 ISA 7.12.2.
// ---------------------------------------------------------------------------
__global__ void k_mlp_wmma(const _Float16* __restrict__ feat,
                           const _Float16* __restrict__ wm16t,
                           const float* __restrict__ bias,
                           float* __restrict__ hout) {
    int tile = blockIdx.x;
    int tm = tile & 7;            // 8 tiles over M=128
    int tn = tile >> 3;           // 20 tiles over N=320
    int lane = threadIdx.x;       // wave32
    int lo = lane & 15, hi = lane >> 4;
    int m = tm * 16 + lo;         // A: M = lane&15 in both lane halves
    int n = tn * 16 + lo;         // B/D: N = lane&15

    v8f acc = {};
    #pragma unroll
    for (int kt = 0; kt < KPAD_ / 32; ++kt) {
        // A 16x32 f16: halves 0..7 -> K = kt*32 + 8*hi + h ; halves 8..15 -> +8
        const _Float16* ap = feat + (size_t)m * KPAD_ + kt * 32 + hi * 8;
        v8h a0 = *(const v8h*)(ap);
        v8h a1 = *(const v8h*)(ap + 16);
        v16h a = __builtin_shufflevector(a0, a1, 0, 1, 2, 3, 4, 5, 6, 7,
                                                 8, 9, 10, 11, 12, 13, 14, 15);
        // B 32x16 f16: halves 0..7 -> K = kt*32 + 8*hi + h ; halves 8..15 -> +8
        // (matches A); transposed storage gives two 16-byte runs.
        const _Float16* bp = wm16t + (size_t)n * KPAD_ + kt * 32 + hi * 8;
        v8h b0 = *(const v8h*)(bp);
        v8h b1 = *(const v8h*)(bp + 16);
        v16h b = __builtin_shufflevector(b0, b1, 0, 1, 2, 3, 4, 5, 6, 7,
                                                 8, 9, 10, 11, 12, 13, 14, 15);
        acc = __builtin_amdgcn_wmma_f32_16x16x32_f16(
            false, a, false, b, (short)0, acc, false, false);
    }
    float bv = bias[n];
    #pragma unroll
    for (int v = 0; v < 8; ++v) {               // D: M = v + 8*hi, N = lane&15
        int mg = tm * 16 + hi * 8 + v;
        hout[(size_t)mg * ATT_ + n] = tanhf(acc[v] + bv);
    }
}

// ---------------------------------------------------------------------------
// Kernel 3: e = h @ w_gvec + b_gvec ; u = softmax(2*e) over channels
// One block per batch; 8 waves each reduce one channel's 320-dot.
// ---------------------------------------------------------------------------
__global__ void k_att(const float* __restrict__ hin, const float* __restrict__ wg,
                      const float* __restrict__ bg, float* __restrict__ u) {
    int b = blockIdx.x;
    int c = threadIdx.x >> 5, lane = threadIdx.x & 31;
    __shared__ float esh[C_];
    float p = 0.0f;
    for (int a = lane; a < ATT_; a += 32)
        p += hin[(size_t)(b * C_ + c) * ATT_ + a] * wg[a];
    #pragma unroll
    for (int off = 16; off > 0; off >>= 1) p += __shfl_down(p, off);
    if (lane == 0) esh[c] = p + bg[0];
    __syncthreads();
    if (threadIdx.x == 0) {
        float mx = -1e30f;
        #pragma unroll
        for (int j = 0; j < C_; ++j) mx = fmaxf(mx, 2.0f * esh[j]);
        float s = 0.0f, ex[C_];
        #pragma unroll
        for (int j = 0; j < C_; ++j) { ex[j] = expf(2.0f * esh[j] - mx); s += ex[j]; }
        #pragma unroll
        for (int j = 0; j < C_; ++j) u[b * C_ + j] = ex[j] / s;
    }
}

// ---------------------------------------------------------------------------
// Kernel 4: MVDR weights. One thread per (b,f):
//   n = (psd_n + EPS*I)*100/temp + reg*1e-5 ; inv = GJ(n) (complex, no pivot)
//   tr = sum_e [inv @ s]_ee ; ws = inv @ (s @ u) / (tr + EPS)
// Fully unrolled 8x8 complex Gauss-Jordan (HPD + I => no pivoting needed).
// Prefetch psd_s rows so their fetch overlaps the inversion ALU work.
// ---------------------------------------------------------------------------
__global__ void k_mvdr(const float* __restrict__ Sr, const float* __restrict__ Si,
                       const float* __restrict__ Nr, const float* __restrict__ Ni,
                       const float* __restrict__ Rr, const float* __restrict__ Ri,
                       const float* __restrict__ u, const unsigned* __restrict__ tmax,
                       float* __restrict__ wsr_out, float* __restrict__ wsi_out) {
    int idx = blockIdx.x * blockDim.x + threadIdx.x;
    if (idx >= BF_) return;
    int b = idx / F_;
    float temp = __uint_as_float(*tmax);
    float scale = 100.0f / temp;
    size_t base = (size_t)idx * 64;

    // Warm the psd_s tile (needed after the inversion) while we do ALU work.
    __builtin_prefetch(Sr + base, 0, 0);
    __builtin_prefetch(Si + base, 0, 0);

    float nr[8][8], ni[8][8];
    #pragma unroll
    for (int i = 0; i < 8; ++i) {
        #pragma unroll
        for (int j = 0; j < 8; ++j) {
            float dg = (i == j) ? EPS_ : 0.0f;
            nr[i][j] = (Nr[base + i * 8 + j] + dg) * scale + Rr[base + i * 8 + j] * 1e-5f;
            ni[i][j] =  Ni[base + i * 8 + j] * scale       + Ri[base + i * 8 + j] * 1e-5f;
        }
    }
    // In-place complex Gauss-Jordan inversion (no pivoting)
    #pragma unroll
    for (int k = 0; k < 8; ++k) {
        float pr = nr[k][k], pi = ni[k][k];
        float den = pr * pr + pi * pi;
        float ipr = pr / den, ipi = -pi / den;
        nr[k][k] = ipr; ni[k][k] = ipi;
        #pragma unroll
        for (int j = 0; j < 8; ++j) {
            if (j == k) continue;
            float r = nr[k][j], im = ni[k][j];
            nr[k][j] = r * ipr - im * ipi;
            ni[k][j] = r * ipi + im * ipr;
        }
        #pragma unroll
        for (int i = 0; i < 8; ++i) {
            if (i == k) continue;
            float fr = nr[i][k], fi = ni[i][k];
            #pragma unroll
            for (int j = 0; j < 8; ++j) {
                if (j == k) continue;
                nr[i][j] -= fr * nr[k][j] - fi * ni[k][j];
                ni[i][j] -= fr * ni[k][j] + fi * nr[k][j];
            }
            nr[i][k] = -(fr * ipr - fi * ipi);
            ni[i][k] = -(fr * ipi + fi * ipr);
        }
    }

    float uu[8];
    #pragma unroll
    for (int d = 0; d < 8; ++d) uu[d] = u[b * C_ + d];

    // Single pass over psd_s rows: v = s @ u and tr = sum_{e,c} inv[e][c]*s[c][e]
    float vr[8], vi[8], trr = 0.0f, tri = 0.0f;
    #pragma unroll
    for (int c = 0; c < 8; ++c) {
        float sr[8], si[8];
        #pragma unroll
        for (int d = 0; d < 8; ++d) {
            sr[d] = Sr[base + c * 8 + d] * scale;
            si[d] = Si[base + c * 8 + d] * scale;
        }
        float ar = 0.f, ai = 0.f;
        #pragma unroll
        for (int d = 0; d < 8; ++d) { ar += sr[d] * uu[d]; ai += si[d] * uu[d]; }
        vr[c] = ar; vi[c] = ai;
        #pragma unroll
        for (int e = 0; e < 8; ++e) {
            trr += nr[e][c] * sr[e] - ni[e][c] * si[e];
            tri += nr[e][c] * si[e] + ni[e][c] * sr[e];
        }
    }
    float drc = trr + EPS_, dic = tri;
    float dd = drc * drc + dic * dic;
    #pragma unroll
    for (int e = 0; e < 8; ++e) {
        float ar = 0.f, ai = 0.f;
        #pragma unroll
        for (int c = 0; c < 8; ++c) {
            ar += nr[e][c] * vr[c] - ni[e][c] * vi[c];
            ai += nr[e][c] * vi[c] + ni[e][c] * vr[c];
        }
        wsr_out[(size_t)idx * 8 + e] = (ar * drc + ai * dic) / dd;
        wsi_out[(size_t)idx * 8 + e] = (ai * drc - ar * dic) / dd;
    }
}

// ---------------------------------------------------------------------------
// Kernel 5: es = conj(ws)^H @ data  -> (B,F,T,2). Pure bandwidth (~300 MB,
// ~13 us floor at 23.3 TB/s). One block per (b,f); 256 threads; each thread
// handles 4 timesteps via b128 loads/stores: 16 loads + 2 stores, all
// coalesced and 16B-aligned.
// ---------------------------------------------------------------------------
__global__ void k_apply(const float* __restrict__ dr, const float* __restrict__ di,
                        const float* __restrict__ wsr, const float* __restrict__ wsi,
                        float* __restrict__ out) {
    int bf = blockIdx.x;
    int t4 = threadIdx.x;                       // 256 threads * 4 t = T
    const float4* xr4 = (const float4*)(dr + (size_t)bf * C_ * T_);
    const float4* xi4 = (const float4*)(di + (size_t)bf * C_ * T_);
    float wr[8], wi[8];
    #pragma unroll
    for (int c = 0; c < 8; ++c) { wr[c] = wsr[bf * 8 + c]; wi[c] = wsi[bf * 8 + c]; }

    float4 er = {0.f, 0.f, 0.f, 0.f}, ei = {0.f, 0.f, 0.f, 0.f};
    #pragma unroll
    for (int c = 0; c < 8; ++c) {
        float4 xr = xr4[c * (T_ / 4) + t4];
        float4 xi = xi4[c * (T_ / 4) + t4];
        er.x = fmaf(wr[c], xr.x, er.x); er.x = fmaf(wi[c], xi.x, er.x);
        er.y = fmaf(wr[c], xr.y, er.y); er.y = fmaf(wi[c], xi.y, er.y);
        er.z = fmaf(wr[c], xr.z, er.z); er.z = fmaf(wi[c], xi.z, er.z);
        er.w = fmaf(wr[c], xr.w, er.w); er.w = fmaf(wi[c], xi.w, er.w);
        ei.x = fmaf(wr[c], xi.x, ei.x); ei.x = fmaf(-wi[c], xr.x, ei.x);
        ei.y = fmaf(wr[c], xi.y, ei.y); ei.y = fmaf(-wi[c], xr.y, ei.y);
        ei.z = fmaf(wr[c], xi.z, ei.z); ei.z = fmaf(-wi[c], xr.z, ei.z);
        ei.w = fmaf(wr[c], xi.w, ei.w); ei.w = fmaf(-wi[c], xr.w, ei.w);
    }
    // Interleave (re,im) pairs: two b128 stores cover 4 timesteps.
    float4* o = (float4*)(out + (size_t)bf * T_ * 2 + (size_t)t4 * 8);
    o[0] = make_float4(er.x, ei.x, er.y, ei.y);
    o[1] = make_float4(er.z, ei.z, er.w, ei.w);
}

// ---------------------------------------------------------------------------
// Workspace layout (bytes, 256-aligned)
//   0      : w_mlp f16 transposed/padded 320*288*2 = 184320
//   184320 : feat f16 padded             128*288*2 =  73728
//   258048 : h f32                       128*320*4 = 163840
//   421888 : u f32                       128*4
//   422400 : ws_r f32                    4112*8*4  = 131584
//   553984 : ws_i f32                    131584
//   685568 : temp (uint)                 4
// ---------------------------------------------------------------------------
extern "C" void kernel_launch(void* const* d_in, const int* in_sizes, int n_in,
                              void* d_out, int out_size, void* d_ws, size_t ws_size,
                              hipStream_t stream) {
    (void)in_sizes; (void)n_in; (void)out_size; (void)ws_size;

    const float* psd_s_r = (const float*)d_in[0];
    const float* psd_s_i = (const float*)d_in[1];
    const float* psd_n_r = (const float*)d_in[2];
    const float* psd_n_i = (const float*)d_in[3];
    const float* data_r  = (const float*)d_in[4];
    const float* data_i  = (const float*)d_in[5];
    const float* reg_r   = (const float*)d_in[6];
    const float* reg_i   = (const float*)d_in[7];
    const float* w_mlp   = (const float*)d_in[8];
    const float* b_mlp   = (const float*)d_in[9];
    const float* w_gvec  = (const float*)d_in[10];
    const float* b_gvec  = (const float*)d_in[11];
    // d_in[12] = ilens: unused by the reference computation

    char* ws = (char*)d_ws;
    _Float16* wm16t = (_Float16*)(ws + 0);
    _Float16* feat  = (_Float16*)(ws + 184320);
    float*    h     = (float*)(ws + 258048);
    float*    u     = (float*)(ws + 421888);
    float*    wsr   = (float*)(ws + 422400);
    float*    wsi   = (float*)(ws + 553984);
    unsigned* tmax  = (unsigned*)(ws + 685568);

    k_conv_wmlp<<<(ATT_ * KPAD_ + 255) / 256, 256, 0, stream>>>(w_mlp, wm16t);
    k_feat<<<(M_ * KPAD_ + 255) / 256, 256, 0, stream>>>(psd_s_r, psd_s_i, feat);
    k_init_temp<<<1, 32, 0, stream>>>(tmax);
    k_absmax<<<256, 256, 0, stream>>>(psd_n_r, BF_ * 64, 1, tmax);
    k_absmax<<<256, 256, 0, stream>>>(psd_n_i, BF_ * 64, 0, tmax);
    k_mlp_wmma<<<160, 32, 0, stream>>>(feat, wm16t, b_mlp, h);
    k_att<<<B_, 256, 0, stream>>>(h, w_gvec, b_gvec, u);
    k_mvdr<<<(BF_ + 127) / 128, 128, 0, stream>>>(psd_s_r, psd_s_i, psd_n_r, psd_n_i,
                                                  reg_r, reg_i, u, tmax, wsr, wsi);
    k_apply<<<BF_, 256, 0, stream>>>(data_r, data_i, wsr, wsi, (float*)d_out);
}
